// LinearEncoderLabelling_29858612641820
// MI455X (gfx1250) — matrix operand order repaired and checked by reading, not compiled, not run on hardware
//
#include <hip/hip_runtime.h>

// ---------------- problem constants (match reference) ----------------
#define BB 16
#define LL 2048
#define DD 512
#define HH 8
#define FFN 2048
#define NLAYERS 6
#define DHH 64
#define MM (BB * LL) // 32768 token rows

// ---------------- types ----------------
typedef __bf16 bf16_t;
typedef bf16_t bf16x8  __attribute__((ext_vector_type(8)));
typedef bf16_t bf16x16 __attribute__((ext_vector_type(16)));
typedef float  f32x8   __attribute__((ext_vector_type(8)));
typedef unsigned int u32x4 __attribute__((ext_vector_type(4)));
typedef int i32x4 __attribute__((ext_vector_type(4)));
typedef int i32x8 __attribute__((ext_vector_type(8)));

__device__ __forceinline__ unsigned short f32_to_bf16_rne(float f) {
  unsigned int u = __builtin_bit_cast(unsigned int, f);
  u += 0x7FFFu + ((u >> 16) & 1u);
  return (unsigned short)(u >> 16);
}
__device__ __forceinline__ bf16_t tobf(float f) {
  return __builtin_bit_cast(bf16_t, f32_to_bf16_rne(f));
}
__device__ __forceinline__ float phi_elu1(float x) { // elu(x)+1
  return x > 0.f ? x + 1.f : __expf(x);
}

// ---------------- Tensor Data Mover: 2-D tile Global -> LDS ----------------
// Builds the D# descriptor (CDNA5 ISA ch.8) and issues tensor_load_to_lds.
// All arguments must be wave-uniform. data_size = 2 bytes (bf16).
// This toolchain exposes the 6-arg builtin:
//   (uint32x4 g0, int32x8 g1, int32x4, int32x4, int32x8, i32 cpol)
__device__ __forceinline__ void tdm_load_2d_bf16(
    unsigned lds_byte_addr, const void* gaddr,
    unsigned tensor_d0,   // elements per row (for OOB)
    unsigned tensor_d1,   // number of rows (for OOB)
    unsigned stride_d0,   // row stride in elements
    unsigned tile_d0,     // tile width (contiguous dim), elements
    unsigned tile_d1)     // tile height (rows)
{
  unsigned long long ga = (unsigned long long)(uintptr_t)gaddr;
  u32x4 g0;
  g0[0] = 1u;                                    // count=1, user descriptor
  g0[1] = lds_byte_addr;                         // lds_addr
  g0[2] = (unsigned)(ga & 0xFFFFFFFFu);          // global_addr[31:0]
  g0[3] = (unsigned)((ga >> 32) & 0x01FFFFFFu)   // global_addr[56:32]
        | (2u << 30);                            // type=2 ("image")
  i32x8 g1;
  g1[0] = (int)(1u << 16);                       // data_size=1 -> 2 bytes; mask=0
  g1[1] = (int)((tensor_d0 & 0xFFFFu) << 16);    // tensor_dim0[15:0] @ [31:16]
  g1[2] = (int)(((tensor_d0 >> 16) & 0xFFFFu) |  // tensor_dim0[31:16]
                ((tensor_d1 & 0xFFFFu) << 16));  // tensor_dim1[15:0]
  g1[3] = (int)(((tensor_d1 >> 16) & 0xFFFFu) |  // tensor_dim1[31:16]
                ((tile_d0 & 0xFFFFu) << 16));    // tile_dim0
  g1[4] = (int)(tile_d1 & 0xFFFFu);              // tile_dim1 (tile_dim2=0)
  g1[5] = (int)stride_d0;                        // tensor_dim0_stride[31:0]
  g1[6] = 0;                                     // stride0[47:32]=0, stride1 lo=0
  g1[7] = 0;                                     // stride1 hi = 0
  i32x4 z4;
  z4[0] = 0; z4[1] = 0; z4[2] = 0; z4[3] = 0;
  i32x8 z8;
  z8[0] = 0; z8[1] = 0; z8[2] = 0; z8[3] = 0;
  z8[4] = 0; z8[5] = 0; z8[6] = 0; z8[7] = 0;
  __builtin_amdgcn_tensor_load_to_lds(g0, g1, z4, z4, z8, 0);
}

// ---------------- embedding + positional encoding ----------------
__global__ __launch_bounds__(256)
void embed_kernel(const int* __restrict__ x, const float* __restrict__ emb,
                  const float* __restrict__ pe, float* __restrict__ h,
                  bf16_t* __restrict__ hb) {
  size_t i = (size_t)blockIdx.x * 256 + threadIdx.x; // < MM*DD
  int bl = (int)(i / DD);
  int d  = (int)(i % DD);
  int l  = bl % LL;
  int tok = x[bl];
  float v = emb[(size_t)tok * DD + d] + pe[(size_t)l * DD + d];
  h[i] = v;
  hb[i] = tobf(v);
}

// ---------------- sequence lengths (mask) ----------------
__global__ __launch_bounds__(256)
void lengths_kernel(const int* __restrict__ x, int* __restrict__ lens) {
  __shared__ int red[256];
  int b = blockIdx.x, t = threadIdx.x, c = 0;
  for (int l = t; l < LL; l += 256) c += (x[(size_t)b * LL + l] != 0) ? 1 : 0;
  red[t] = c; __syncthreads();
  for (int s = 128; s > 0; s >>= 1) { if (t < s) red[t] += red[t + s]; __syncthreads(); }
  if (t == 0) lens[b] = red[0];
}

// ---------------- fp32 W[K][N] -> bf16 Wt[N][K] ----------------
__global__ __launch_bounds__(256)
void transp_bf16_kernel(const float* __restrict__ W, bf16_t* __restrict__ Wt,
                        int Kr, int Nc) {
  size_t i = (size_t)blockIdx.x * 256 + threadIdx.x;
  size_t total = (size_t)Kr * Nc;
  if (i >= total) return;
  int kr = (int)(i % Kr);
  int n  = (int)(i / Kr);
  Wt[(size_t)n * Kr + kr] = tobf(W[(size_t)kr * Nc + n]);
}

// ---------------- bf16 WMMA GEMM with TDM double-buffered LDS staging ----------------
// C[M,N] = A[M,K] @ Bt[N,K]^T + bias.  256 threads = 8 waves.
// Block tile 128(M) x 64(N), K chunks of 64.  Per chunk the TDM DMAs
// A(128x64, 16KB) and B(64x64, 8KB) tiles into LDS (one issue, wave 0 only,
// double buffered, tracked with TENSORcnt).  Waves read fragments via ds_load
// and run v_wmma_f32_16x16x32_bf16.  Dynamic LDS only -> descriptor lds_addr
// offsets are raw group-segment offsets.
#define GEMM_LDS_BYTES 49152
#define LDS_A0 0u
#define LDS_B0 16384u
#define LDS_A1 24576u
#define LDS_B1 40960u

template <int RELU, int HAS_RES, int OUT_F32, int OUT_BF16>
__global__ __launch_bounds__(256)
void gemm_bf16_wmma(const bf16_t* __restrict__ A, const bf16_t* __restrict__ Bt,
                    const float* __restrict__ bias, const float* __restrict__ Res,
                    float* __restrict__ Cf, bf16_t* __restrict__ Cb,
                    int N, int K) {
  extern __shared__ char smem[];
  const int lane = threadIdx.x & 31;
  const int wave = threadIdx.x >> 5;
  const int m0 = blockIdx.y * 128;
  const int n0 = blockIdx.x * 64;

  const int nchunks = K >> 6;
  const bool issuer = (threadIdx.x < 32); // wave 0 issues TDM ops

  if (issuer) {
    tdm_load_2d_bf16(LDS_A0, A + (size_t)m0 * K, K, MM, K, 64u, 128u);
    tdm_load_2d_bf16(LDS_B0, Bt + (size_t)n0 * K, K, (unsigned)N, K, 64u, 64u);
  }

  f32x8 acc[4];
#pragma unroll
  for (int j = 0; j < 4; ++j)
#pragma unroll
    for (int r = 0; r < 8; ++r) acc[j][r] = 0.f;

  // Fragment addressing (contiguous k order: lane half h holds k = 16h + s in both A and B)
  const int arow = wave * 16 + (lane & 15); // A row within block tile
  const int ak = (lane >> 4) * 16;          // A K half
  const int bcol = lane & 15;               // B column within 16-wide tile
  const int bk = (lane >> 4) * 16;          // B K half

  for (int kc = 0; kc < nchunks; ++kc) {
    const int buf = kc & 1;
    if (issuer) {
      if (kc + 1 < nchunks) {
        const unsigned la = buf ? LDS_A0 : LDS_A1;
        const unsigned lb = buf ? LDS_B0 : LDS_B1;
        tdm_load_2d_bf16(la, A + (size_t)m0 * K + (kc + 1) * 64, K, MM, K, 64u, 128u);
        tdm_load_2d_bf16(lb, Bt + (size_t)n0 * K + (kc + 1) * 64, K, (unsigned)N, K, 64u, 64u);
        __builtin_amdgcn_s_wait_tensorcnt((short)2); // chunk kc complete (in-order)
      } else {
        __builtin_amdgcn_s_wait_tensorcnt((short)0);
      }
    }
    __syncthreads(); // tile kc visible to all waves

    const bf16_t* Ab = (const bf16_t*)(smem + (buf ? LDS_A1 : LDS_A0));
    const bf16_t* Bb = (const bf16_t*)(smem + (buf ? LDS_B1 : LDS_B0));
#pragma unroll
    for (int ks = 0; ks < 2; ++ks) {
      const bf16_t* ap = Ab + arow * 64 + ks * 32 + ak;
      bf16x8 alo = *(const bf16x8*)ap;
      bf16x8 ahi = *(const bf16x8*)(ap + 8);
      bf16x16 afrag;
#pragma unroll
      for (int i = 0; i < 8; ++i) { afrag[i] = alo[i]; afrag[8 + i] = ahi[i]; }
#pragma unroll
      for (int j = 0; j < 4; ++j) {
        const bf16_t* bp = Bb + (j * 16 + bcol) * 64 + ks * 32 + bk;
        bf16x16 bfrag = *(const bf16x16*)bp;
        acc[j] = __builtin_amdgcn_wmma_f32_16x16x32_bf16(
            false, afrag, false, bfrag, (short)0, acc[j], false, false);
      }
    }
    __syncthreads(); // before buffer is overwritten two chunks later
  }

  // Epilogue (C/D layout: lane&15 = N; VGPR r -> M = r + 8*(lane>>4))
  const int cm = m0 + wave * 16 + 8 * (lane >> 4);
#pragma unroll
  for (int j = 0; j < 4; ++j) {
    const int n = n0 + j * 16 + (lane & 15);
    const float bv = bias[n];
#pragma unroll
    for (int r = 0; r < 8; ++r) {
      size_t idx = (size_t)(cm + r) * N + n;
      float v = acc[j][r] + bv;
      if (RELU)    v = v > 0.f ? v : 0.f;
      if (HAS_RES) v += Res[idx];
      if (OUT_F32)  Cf[idx] = v;
      if (OUT_BF16) Cb[idx] = tobf(v);
    }
  }
}

// ---------------- phi(Q) in place ----------------
__global__ __launch_bounds__(256)
void phi_q_kernel(float* __restrict__ q) {
  size_t i = (size_t)blockIdx.x * 256 + threadIdx.x;
  q[i] = phi_elu1(q[i]);
}

// ---------------- KV[b,h,m,d] = sum_s phi(K)[b,s,h,d]*V[b,s,h,m]; Ksum[b,h,d] ----------------
__global__ __launch_bounds__(256)
void phi_kv_kernel(const float* __restrict__ Kb, const float* __restrict__ Vb,
                   const int* __restrict__ lens, float* __restrict__ KV,
                   float* __restrict__ Ks) {
  __shared__ float kl[DHH], vl[DHH];
  const int bh = blockIdx.x;          // b*H + h
  const int b = bh / HH, hh = bh % HH;
  const int t = threadIdx.x;
  const int m  = t >> 2;              // 0..63 (v dim)
  const int d0 = (t & 3) * 16;        // k-dim sub-block
  float acc[16];
#pragma unroll
  for (int i = 0; i < 16; ++i) acc[i] = 0.f;
  float ksacc = 0.f;
  const int len = lens[b];
  for (int s = 0; s < LL; ++s) {
    const size_t base = (((size_t)b * LL + s) * HH + hh) * DHH;
    if (t < DHH)            kl[t]        = (s < len) ? phi_elu1(Kb[base + t]) : 0.f;
    else if (t < 2 * DHH)   vl[t - DHH]  = Vb[base + (t - DHH)];
    __syncthreads();
    const float vm = vl[m];
#pragma unroll
    for (int i = 0; i < 16; ++i) acc[i] += vm * kl[d0 + i];
    if (t < DHH) ksacc += kl[t];
    __syncthreads();
  }
#pragma unroll
  for (int i = 0; i < 16; ++i)
    KV[((size_t)bh * DHH + m) * DHH + d0 + i] = acc[i];
  if (t < DHH) Ks[(size_t)bh * DHH + t] = ksacc;
}

// ---------------- out[l,h,m] = Z * sum_d Qp[l,h,d]*KV[b,h,m,d] ----------------
__global__ __launch_bounds__(256)
void attn_out_kernel(const float* __restrict__ Qp, const float* __restrict__ KV,
                     const float* __restrict__ Ks, float* __restrict__ Of,
                     bf16_t* __restrict__ Ob) {
  __shared__ float ql[DD];
  __shared__ float zl[HH];
  const int row = blockIdx.x;       // b*L + l
  const int b = row / LL;
  const int t = threadIdx.x;
  ql[t]       = Qp[(size_t)row * DD + t];
  ql[t + 256] = Qp[(size_t)row * DD + t + 256];
  __syncthreads();
  if (t < HH) {
    const float* ks = Ks + ((size_t)b * HH + t) * DHH;
    float s = 0.f;
    for (int d = 0; d < DHH; ++d) s += ql[t * DHH + d] * ks[d];
    zl[t] = 1.f / (s + 1e-6f);
  }
  __syncthreads();
#pragma unroll
  for (int it = 0; it < 2; ++it) {
    const int idx = t + it * 256;
    const int hh = idx >> 6, m = idx & 63;
    const float* kvp = KV + (((size_t)b * HH + hh) * DHH + m) * DHH;
    float s = 0.f;
    for (int d = 0; d < DHH; ++d) s += ql[hh * DHH + d] * kvp[d];
    const float o = s * zl[hh];
    Of[(size_t)row * DD + idx] = o;
    Ob[(size_t)row * DD + idx] = tobf(o);
  }
}

// ---------------- row LayerNorm over D=512 ----------------
__global__ __launch_bounds__(256)
void layernorm_kernel(const float* __restrict__ X, const float* __restrict__ g,
                      const float* __restrict__ bta, float* __restrict__ Yf,
                      bf16_t* __restrict__ Yb) {
  __shared__ float red[256];
  const int row = blockIdx.x;
  const int t = threadIdx.x;
  const float x0 = X[(size_t)row * DD + t];
  const float x1 = X[(size_t)row * DD + t + 256];
  red[t] = x0 + x1; __syncthreads();
  for (int s = 128; s > 0; s >>= 1) { if (t < s) red[t] += red[t + s]; __syncthreads(); }
  const float mean = red[0] * (1.f / DD);
  __syncthreads();
  const float d0 = x0 - mean, d1 = x1 - mean;
  red[t] = d0 * d0 + d1 * d1; __syncthreads();
  for (int s = 128; s > 0; s >>= 1) { if (t < s) red[t] += red[t + s]; __syncthreads(); }
  const float rstd = rsqrtf(red[0] * (1.f / DD) + 1e-5f);
  const float y0 = d0 * rstd * g[t] + bta[t];
  const float y1 = d1 * rstd * g[t + 256] + bta[t + 256];
  Yf[(size_t)row * DD + t] = y0;
  Yf[(size_t)row * DD + t + 256] = y1;
  if (Yb) {
    Yb[(size_t)row * DD + t] = tobf(y0);
    Yb[(size_t)row * DD + t + 256] = tobf(y1);
  }
}

// ---------------- host-side orchestration ----------------
extern "C" void kernel_launch(void* const* d_in, const int* in_sizes, int n_in,
                              void* d_out, int out_size, void* d_ws, size_t ws_size,
                              hipStream_t stream) {
  (void)in_sizes; (void)n_in; (void)out_size; (void)ws_size;
  const int*   x   = (const int*)d_in[0];
  const float* emb = (const float*)d_in[1];
  const float* pe  = (const float*)d_in[2];
  auto P = [&](int layer, int slot) -> const float* {
    return (const float*)d_in[3 + layer * 16 + slot];
  };
  const float* fin_g = (const float*)d_in[3 + NLAYERS * 16 + 0];
  const float* fin_b = (const float*)d_in[3 + NLAYERS * 16 + 1];

  // workspace carve
  char* ws = (char*)d_ws;
  size_t off = 0;
  auto carve = [&](size_t bytes) -> char* {
    char* p = ws + off;
    off = (off + bytes + 255) & ~(size_t)255;
    return p;
  };
  float*  h    = (float*) carve((size_t)MM * DD * 4);
  bf16_t* hb   = (bf16_t*)carve((size_t)MM * DD * 2);
  float*  q    = (float*) carve((size_t)MM * DD * 4);   // also reused as y2
  float*  kk   = (float*) carve((size_t)MM * DD * 4);
  float*  vv   = (float*) carve((size_t)MM * DD * 4);
  float*  aout = (float*) carve((size_t)MM * DD * 4);
  bf16_t* aoutb= (bf16_t*)carve((size_t)MM * DD * 2);
  bf16_t* ffb  = (bf16_t*)carve((size_t)MM * FFN * 2);
  float*  kvb  = (float*) carve((size_t)BB * HH * DHH * DHH * 4);
  float*  ksum = (float*) carve((size_t)BB * HH * DHH * 4);
  int*    lens = (int*)   carve((size_t)BB * 4);
  bf16_t* wq   = (bf16_t*)carve((size_t)DD * DD * 2);
  bf16_t* wk   = (bf16_t*)carve((size_t)DD * DD * 2);
  bf16_t* wv   = (bf16_t*)carve((size_t)DD * DD * 2);
  bf16_t* wo   = (bf16_t*)carve((size_t)DD * DD * 2);
  bf16_t* w1t  = (bf16_t*)carve((size_t)DD * FFN * 2);
  bf16_t* w2t  = (bf16_t*)carve((size_t)FFN * DD * 2);

  const int elemBlocks = (MM * DD) / 256;           // 65536
  embed_kernel<<<elemBlocks, 256, 0, stream>>>(x, emb, pe, h, hb);
  lengths_kernel<<<BB, 256, 0, stream>>>(x, lens);

  const dim3 gD(DD / 64, MM / 128);    // N=512 GEMMs
  const dim3 gF(FFN / 64, MM / 128);   // N=2048 GEMM

  for (int l = 0; l < NLAYERS; ++l) {
    // weight transpose+convert (fp32 KxN -> bf16 NxK)
    transp_bf16_kernel<<<(DD * DD) / 256, 256, 0, stream>>>(P(l, 0),  wq,  DD,  DD);
    transp_bf16_kernel<<<(DD * DD) / 256, 256, 0, stream>>>(P(l, 2),  wk,  DD,  DD);
    transp_bf16_kernel<<<(DD * DD) / 256, 256, 0, stream>>>(P(l, 4),  wv,  DD,  DD);
    transp_bf16_kernel<<<(DD * DD) / 256, 256, 0, stream>>>(P(l, 6),  wo,  DD,  DD);
    transp_bf16_kernel<<<(DD * FFN) / 256, 256, 0, stream>>>(P(l, 10), w1t, DD,  FFN);
    transp_bf16_kernel<<<(FFN * DD) / 256, 256, 0, stream>>>(P(l, 12), w2t, FFN, DD);

    // Q/K/V projections
    gemm_bf16_wmma<0,0,1,0><<<gD, 256, GEMM_LDS_BYTES, stream>>>(hb, wq, P(l, 1), nullptr, q,  nullptr, DD, DD);
    gemm_bf16_wmma<0,0,1,0><<<gD, 256, GEMM_LDS_BYTES, stream>>>(hb, wk, P(l, 3), nullptr, kk, nullptr, DD, DD);
    gemm_bf16_wmma<0,0,1,0><<<gD, 256, GEMM_LDS_BYTES, stream>>>(hb, wv, P(l, 5), nullptr, vv, nullptr, DD, DD);

    // linear attention core
    phi_q_kernel<<<elemBlocks, 256, 0, stream>>>(q);
    phi_kv_kernel<<<BB * HH, 256, 0, stream>>>(kk, vv, lens, kvb, ksum);
    attn_out_kernel<<<MM, 256, 0, stream>>>(q, kvb, ksum, aout, aoutb);

    // output projection + residual: h = h + attn @ Wo + bo
    gemm_bf16_wmma<0,1,1,0><<<gD, 256, GEMM_LDS_BYTES, stream>>>(aoutb, wo, P(l, 7), h, h, nullptr, DD, DD);
    // LN1 -> h (f32) + hb (bf16)
    layernorm_kernel<<<MM, 256, 0, stream>>>(h, P(l, 8), P(l, 9), h, hb);
    // FFN: relu(hb @ W1 + b1) -> bf16
    gemm_bf16_wmma<1,0,0,1><<<gF, 256, GEMM_LDS_BYTES, stream>>>(hb, w1t, P(l, 11), nullptr, nullptr, ffb, FFN, DD);
    // y2 = ffb @ W2 + b2 + h  (into q buffer)
    gemm_bf16_wmma<0,1,1,0><<<gD, 256, GEMM_LDS_BYTES, stream>>>(ffb, w2t, P(l, 13), h, q, nullptr, DD, FFN);
    // LN2 -> h + hb
    layernorm_kernel<<<MM, 256, 0, stream>>>(q, P(l, 14), P(l, 15), h, hb);
  }

  // final layer norm -> d_out (fp32)
  layernorm_kernel<<<MM, 256, 0, stream>>>(h, fin_g, fin_b, (float*)d_out, nullptr);
}
